// AttnGraphConvolution_46746424049692
// MI455X (gfx1250) — matrix operand order repaired and checked by reading, not compiled, not run on hardware
//
#include <hip/hip_runtime.h>
#include <hip/hip_bf16.h>

#define NN     12288
#define INF_   128
#define OF     64
#define ALPHA_ 0.2f
#define NEGINF (-9e15f)
#define NTILES 768      // NN/16 row tiles
#define JCH    8        // j-dimension split factor (concurrency for HBM)
#define JSPAN  (NN / JCH)

typedef __attribute__((ext_vector_type(16))) __bf16 v16bf;
typedef __attribute__((ext_vector_type(8)))  __bf16 v8bf;
typedef __attribute__((ext_vector_type(8)))  float  v8f;
typedef __attribute__((ext_vector_type(4)))  int    v4i;
typedef __attribute__((ext_vector_type(4)))  float  v4f;

// ---------------------------------------------------------------------------
// K1: h = input @ W; store h TRANSPOSED in bf16 (hbT[64][12288]) so that the
// WMMA B-operand (K contiguous per lane) is two global_load_b128 from L2.
// Also Wh1 = h.a1, Wh2 = h.a2 via LDS reduction. One row per block, 64 thr.
// ---------------------------------------------------------------------------
__global__ void gat_linear_kernel(const float* __restrict__ in,
                                  const float* __restrict__ W,
                                  const float* __restrict__ a,
                                  __bf16* __restrict__ hbT,
                                  float* __restrict__ wh1,
                                  float* __restrict__ wh2) {
  const int i = blockIdx.x;
  const int t = threadIdx.x;  // 0..63
  __shared__ float inL[INF_];
  __shared__ float r1[OF];
  __shared__ float r2[OF];
  inL[t]      = in[i * INF_ + t];
  inL[t + 64] = in[i * INF_ + t + 64];
  __syncthreads();
  float acc = 0.f;
#pragma unroll 8
  for (int k = 0; k < INF_; ++k) acc = fmaf(inL[k], W[k * OF + t], acc);
  hbT[(size_t)t * NN + i] = (__bf16)acc;  // transposed store (1.5MB once)
  r1[t] = acc * a[t];
  r2[t] = acc * a[OF + t];
  __syncthreads();
#pragma unroll
  for (int s = 32; s > 0; s >>= 1) {
    if (t < s) { r1[t] += r1[t + s]; r2[t] += r2[t + s]; }
    __syncthreads();
  }
  if (t == 0) { wh1[i] = r1[0]; wh2[i] = r2[0]; }
}

// ---------------------------------------------------------------------------
// K2: per-row online softmax statistics (max m, sum s); masked entries are
// NEG_INF-valued terms, exactly matching reference (all-masked row -> s = N).
// adj read with NT hint (single-use 604MB stream must not wash L2).
// ---------------------------------------------------------------------------
__global__ void gat_stats_kernel(const int* __restrict__ adj,
                                 const float* __restrict__ wh1,
                                 const float* __restrict__ wh2,
                                 float* __restrict__ mOut,
                                 float* __restrict__ sOut) {
  const int row = blockIdx.x;
  const int t = threadIdx.x;  // 0..255
  const float w1 = wh1[row];
  const int* arow = adj + (long long)row * NN;
  float m = -__builtin_inff();
  float s = 0.f;
  for (int c = t; c < NN; c += 256) {
    int aval = __builtin_nontemporal_load(arow + c);
    float u = w1 + wh2[c];
    float e = fmaxf(u, ALPHA_ * u);        // leaky-relu, branchless
    float ev = (aval > 0) ? e : NEGINF;    // value select only
    if (ev > m) {
      s = s * __expf(m - ev) + 1.f;
      m = ev;
    } else {
      s += __expf(ev - m);
    }
  }
  __shared__ float mS[256];
  __shared__ float sS[256];
  mS[t] = m; sS[t] = s;
  __syncthreads();
#pragma unroll
  for (int off = 128; off > 0; off >>= 1) {
    if (t < off) {
      float m1 = mS[t], s1 = sS[t];
      float m2 = mS[t + off], s2 = sS[t + off];
      float M = fmaxf(m1, m2);
      sS[t] = s1 * __expf(m1 - M) + s2 * __expf(m2 - M);
      mS[t] = M;
    }
    __syncthreads();
  }
  if (t == 0) { mOut[row] = mS[0]; sOut[row] = sS[0]; }
}

// ---------------------------------------------------------------------------
// K3: fused probabilities + (P @ h) via v_wmma_f32_16x16x32_bf16, no LDS.
// Grid: (768 row-tiles) x (8 j-chunks); one wave per block; each block emits
// a 16x64 f32 partial accumulator (un-normalized).
// A fragment (16x32): lane = hi*16 + m; elem e -> K = hi*16 + e  (contiguous)
// B fragment (32x16): lane = hi*16 + n; elem e -> K = hi*16 + e  (contiguous)
// C/D (16x16 f32):    VGPR r -> M = r + hi*8, N = lane&15
// ---------------------------------------------------------------------------
__global__ void gat_attn_wmma_kernel(const int* __restrict__ adj,
                                     const __bf16* __restrict__ hbT,
                                     const float* __restrict__ wh1G,
                                     const float* __restrict__ wh2G,
                                     const float* __restrict__ mG,
                                     float* __restrict__ part) {
  const int l    = threadIdx.x;        // 0..31
  const int bt   = blockIdx.x;         // row tile
  const int bc   = blockIdx.y;         // j chunk
  const int row0 = bt * 16;
  const int hi   = l >> 4;
  const int ml   = l & 15;

  const float w1     = wh1G[row0 + ml];
  const float mrow   = mG[row0 + ml];
  const float negArg = NEGINF - mrow;  // hoisted masked-entry exp argument

  v8f acc0 = {}, acc1 = {}, acc2 = {}, acc3 = {};

  const long long arowBase = (long long)(row0 + ml) * NN;
  const int kbase = hi * 8;
  const int jlo = bc * JSPAN;
  const int jhi = jlo + JSPAN;

  for (int j = jlo; j < jhi; j += 32) {
    // unconditional stream prefetch, 8 iterations ahead (global_prefetch_b8)
    __builtin_prefetch(adj + arowBase + j + 256, 0, 1);

    const v4i* ap = (const v4i*)(adj + arowBase + j);
    const v4f* wp = (const v4f*)(wh2G + j);

    // --- probabilities straight into the A-fragment layout, branchless ---
    v16bf aF;
#pragma unroll
    for (int q = 0; q < 4; ++q) {
      const int chunk = (hi * 16 + q * 4) >> 2;  // v4 idx
      v4i av = __builtin_nontemporal_load(ap + chunk);  // NT: single-use stream
      v4f wv = wp[chunk];                               // RT: L2-resident
      float u0 = w1 + wv[0], u1 = w1 + wv[1], u2 = w1 + wv[2], u3 = w1 + wv[3];
      float e0 = fmaxf(u0, ALPHA_ * u0) - mrow;
      float e1 = fmaxf(u1, ALPHA_ * u1) - mrow;
      float e2 = fmaxf(u2, ALPHA_ * u2) - mrow;
      float e3 = fmaxf(u3, ALPHA_ * u3) - mrow;
      aF[q * 4 + 0] = (__bf16)__expf((av[0] > 0) ? e0 : negArg);
      aF[q * 4 + 1] = (__bf16)__expf((av[1] > 0) ? e1 : negArg);
      aF[q * 4 + 2] = (__bf16)__expf((av[2] > 0) ? e2 : negArg);
      aF[q * 4 + 3] = (__bf16)__expf((av[3] > 0) ? e3 : negArg);
    }

    // --- B fragments: two b128 loads each from transposed h (L2-resident) ---
#define GAT_DO_TILE(ACC, T)                                                    \
    {                                                                          \
      const uint4* bp = (const uint4*)(hbT +                                   \
          (size_t)((T) * 16 + ml) * NN + j + hi * 16);                         \
      v8bf blo = __builtin_bit_cast(v8bf, bp[0]);                              \
      v8bf bhi = __builtin_bit_cast(v8bf, bp[1]);                              \
      v16bf bF = __builtin_shufflevector(blo, bhi, 0, 1, 2, 3, 4, 5, 6, 7,     \
                                         8, 9, 10, 11, 12, 13, 14, 15);        \
      ACC = __builtin_amdgcn_wmma_f32_16x16x32_bf16(                           \
          false, aF, false, bF, (short)0, ACC, false, false);                  \
    }
    GAT_DO_TILE(acc0, 0)
    GAT_DO_TILE(acc1, 1)
    GAT_DO_TILE(acc2, 2)
    GAT_DO_TILE(acc3, 3)
#undef GAT_DO_TILE
  }

  // --- write 16x64 partial tile (row-major, NT: write-once scratch) ---
  float* pOut = part + ((size_t)bc * NTILES + bt) * (16 * OF);
#define GAT_STORE_TILE(ACC, T)                                                 \
  {                                                                            \
    _Pragma("unroll")                                                          \
    for (int r = 0; r < 8; ++r)                                                \
      __builtin_nontemporal_store(ACC[r],                                      \
          pOut + (r + hi * 8) * OF + (T) * 16 + ml);                           \
  }
  GAT_STORE_TILE(acc0, 0)
  GAT_STORE_TILE(acc1, 1)
  GAT_STORE_TILE(acc2, 2)
  GAT_STORE_TILE(acc3, 3)
#undef GAT_STORE_TILE
}

// ---------------------------------------------------------------------------
// K4: sum the 8 j-chunk partials, scale by 1/rowsum, ELU, store output.
// 768 blocks x 256 threads, fully coalesced.
// ---------------------------------------------------------------------------
__global__ void gat_reduce_kernel(const float* __restrict__ part,
                                  const float* __restrict__ sG,
                                  float* __restrict__ out) {
  const int b = blockIdx.x;   // row tile
  const int t = threadIdx.x;  // 0..255
#pragma unroll
  for (int q = 0; q < 4; ++q) {
    const int idx = q * 256 + t;  // 0..1023 within 16x64 tile
    float v = 0.f;
#pragma unroll
    for (int c = 0; c < JCH; ++c)
      v += __builtin_nontemporal_load(
          part + ((size_t)c * NTILES + b) * (16 * OF) + idx);
    const int row = b * 16 + (idx >> 6);
    v *= (1.f / sG[row]);
    v = (v > 0.f) ? v : (__expf(v) - 1.f);
    out[(size_t)b * (16 * OF) + idx] = v;
  }
}

// ---------------------------------------------------------------------------
extern "C" void kernel_launch(void* const* d_in, const int* in_sizes, int n_in,
                              void* d_out, int out_size, void* d_ws, size_t ws_size,
                              hipStream_t stream) {
  (void)in_sizes; (void)n_in; (void)out_size; (void)ws_size;
  const float* input = (const float*)d_in[0];  // (N, 128) f32
  const int*   adj   = (const int*)d_in[1];    // (N, N) i32
  const float* W     = (const float*)d_in[2];  // (128, 64) f32
  const float* a     = (const float*)d_in[3];  // (128, 1) f32
  float*       out   = (float*)d_out;          // (N, 64) f32

  char* ws = (char*)d_ws;
  __bf16* hbT  = (__bf16*)(ws);                 // 64*12288*2 = 1,572,864 B
  float*  wh1  = (float*)(ws + 1572864);        // 49,152 B
  float*  wh2  = (float*)(ws + 1622016);        // 49,152 B
  float*  mrow = (float*)(ws + 1671168);        // 49,152 B
  float*  srow = (float*)(ws + 1720320);        // 49,152 B
  float*  part = (float*)(ws + 1769472);        // 8*768*1024*4 = 25,165,824 B

  gat_linear_kernel<<<NN, 64, 0, stream>>>(input, W, a, hbT, wh1, wh2);
  gat_stats_kernel<<<NN, 256, 0, stream>>>(adj, wh1, wh2, mrow, srow);
  dim3 g3(NTILES, JCH);
  gat_attn_wmma_kernel<<<g3, 32, 0, stream>>>(adj, hbT, wh1, wh2, mrow, part);
  gat_reduce_kernel<<<NTILES, 256, 0, stream>>>(part, srow, out);
}
